// SelectiveSSM_74045236183625
// MI455X (gfx1250) — matrix operand order repaired and checked by reading, not compiled, not run on hardware
//
#include <hip/hip_runtime.h>
#include <hip/hip_bf16.h>

typedef __attribute__((ext_vector_type(16))) _Float16 v16h;
typedef __attribute__((ext_vector_type(8)))  _Float16 v8h;
typedef __attribute__((ext_vector_type(8)))  float    v8f;

#define BATCH    2
#define SEQ      1024
#define D_MODEL  1024
#define D_INNER  2048
#define D_STATE  16
#define NROWS    (BATCH * SEQ)          // 2048

// ---------------- GEMM: C[M,N] = A[M,K] * W[N,K]^T, f16 in, f32 out ---------
#define BM 128
#define BN 128
#define BK 32
#define LDSS 40   // padded f16 stride (80B rows, 16B-aligned)

__device__ inline v16h cat8(v8h lo, v8h hi) {
  return __builtin_shufflevector(lo, hi, 0,1,2,3,4,5,6,7,8,9,10,11,12,13,14,15);
}

// A fragment (16x32 f16), spec 7.12.2
__device__ inline v16h load_frag_a(const _Float16* as, int m0, int lane) {
  const int row = lane & 15;
  const int kh  = (lane >> 4) << 3;            // 0 or 8
  const _Float16* p = as + (m0 + row) * LDSS + kh;
  v8h lo = *(const v8h*)(p);                   // K = kh .. kh+7
  v8h hi = *(const v8h*)(p + 16);              // K = 16+kh .. 16+kh+7
  return cat8(lo, hi);
}

// B fragment (32x16 f16): lane&15 = N column; lanes<16: K=0..7,16..23, lanes>=16: K=8..15,24..31
__device__ inline v16h load_frag_b(const _Float16* bs, int n0, int lane) {
  const int col = lane & 15;
  const int kh  = (lane >> 4) << 3;            // 0 or 8
  const _Float16* p = bs + (n0 + col) * LDSS + kh;
  v8h lo = *(const v8h*)(p);
  v8h hi = *(const v8h*)(p + 16);
  return cat8(lo, hi);
}

// CDNA5 async global->LDS copy (ASYNCcnt-tracked), 16B per lane.
// LDS address = low 32 bits of generic shared pointer (flat aperture mapping).
__device__ __forceinline__ void async_copy_b128(_Float16* lp, const _Float16* gp) {
  unsigned lds = (unsigned)(uintptr_t)lp;
  unsigned long long ga = (unsigned long long)(uintptr_t)gp;
  asm volatile("global_load_async_to_lds_b128 %0, %1, off"
               :: "v"(lds), "v"(ga) : "memory");
}
__device__ __forceinline__ void wait_async_le4() {
  asm volatile("s_wait_asynccnt 0x4" ::: "memory");
}
__device__ __forceinline__ void wait_async_0() {
  asm volatile("s_wait_asynccnt 0x0" ::: "memory");
}

template<int NDIM, int KDIM>
__launch_bounds__(256)
__global__ void gemm_f16_wmma(const _Float16* __restrict__ A,
                              const _Float16* __restrict__ W,
                              float* __restrict__ C) {
  __shared__ alignas(16) _Float16 a_s[2][BM * LDSS];
  __shared__ alignas(16) _Float16 b_s[2][BN * LDSS];

  const int tid  = threadIdx.x;
  const int lane = tid & 31;
  const int wave = tid >> 5;                   // 0..7
  const int m_block = blockIdx.y * BM;
  const int n_block = blockIdx.x * BN;
  const int wm = (wave & 3) * 32;              // wave M offset (2 subtiles)
  const int wn = (wave >> 2) * 64;             // wave N offset (4 subtiles)

  v8f acc[2][4];
  for (int mi = 0; mi < 2; ++mi)
    for (int ni = 0; ni < 4; ++ni)
      for (int e = 0; e < 8; ++e) acc[mi][ni][e] = 0.0f;

  const int lrow = tid >> 1;                   // 0..127
  const int lkc  = (tid & 1) * 16;             // 0 or 16

  const _Float16* gA = A + (size_t)(m_block + lrow) * KDIM + lkc;
  const _Float16* gB = W + (size_t)(n_block + lrow) * KDIM + lkc;
  const int lds_off = lrow * LDSS + lkc;

  // prologue: stage tile 0 into buffer 0 (4 async b128 per thread)
  async_copy_b128(&a_s[0][lds_off],     gA);
  async_copy_b128(&a_s[0][lds_off + 8], gA + 8);
  async_copy_b128(&b_s[0][lds_off],     gB);
  async_copy_b128(&b_s[0][lds_off + 8], gB + 8);

  int buf = 0;
  for (int k0 = 0; k0 < KDIM; k0 += BK) {
    __syncthreads();                           // all readers of buf^1 done
    if (k0 + BK < KDIM) {
      const int nb = buf ^ 1;
      async_copy_b128(&a_s[nb][lds_off],     gA + k0 + BK);
      async_copy_b128(&a_s[nb][lds_off + 8], gA + k0 + BK + 8);
      async_copy_b128(&b_s[nb][lds_off],     gB + k0 + BK);
      async_copy_b128(&b_s[nb][lds_off + 8], gB + k0 + BK + 8);
      wait_async_le4();                        // current tile's 4 ops complete
    } else {
      wait_async_0();
    }
    __syncthreads();                           // everyone's tile landed in LDS

    v16h af[2];
    af[0] = load_frag_a(a_s[buf], wm,      lane);
    af[1] = load_frag_a(a_s[buf], wm + 16, lane);
    v16h bf[4];
    #pragma unroll
    for (int ni = 0; ni < 4; ++ni) bf[ni] = load_frag_b(b_s[buf], wn + ni * 16, lane);

    #pragma unroll
    for (int mi = 0; mi < 2; ++mi)
      #pragma unroll
      for (int ni = 0; ni < 4; ++ni)
        acc[mi][ni] = __builtin_amdgcn_wmma_f32_16x16x32_f16(
            false, af[mi], false, bf[ni], (short)0, acc[mi][ni], false, false);
    buf ^= 1;
  }

  // C/D layout: VGPR v -> row (lane>=16 ? 8 : 0)+v, col = lane&15
  const int col    = lane & 15;
  const int rowoff = (lane >> 4) * 8;
  for (int mi = 0; mi < 2; ++mi)
    for (int ni = 0; ni < 4; ++ni) {
      const int mbase = m_block + wm + mi * 16 + rowoff;
      const int nbase = n_block + wn + ni * 16 + col;
      #pragma unroll
      for (int v = 0; v < 8; ++v)
        C[(size_t)(mbase + v) * NDIM + nbase] = acc[mi][ni][v];
    }
}

// ---------------- f32 -> f16 conversion ------------------------------------
__global__ void cvt_f16_kernel(const float* __restrict__ in, _Float16* __restrict__ out, int n) {
  int i = blockIdx.x * blockDim.x + threadIdx.x;
  if (i < n) out[i] = (_Float16)in[i];
}

// ---------------- depthwise causal conv (D_CONV=4) + SiLU -------------------
__global__ void conv_silu_kernel(const float* __restrict__ xz,
                                 const float* __restrict__ cw,
                                 const float* __restrict__ cb,
                                 float* __restrict__ xc) {
  int idx = blockIdx.x * blockDim.x + threadIdx.x;   // b*S*DI + t*DI + d
  if (idx >= BATCH * SEQ * D_INNER) return;
  const int d = idx & (D_INNER - 1);
  const int t = (idx >> 11) & (SEQ - 1);
  const int b = idx >> 21;
  float acc = cb[d];
  #pragma unroll
  for (int j = 0; j < 4; ++j) {
    int tt = t - 3 + j;
    if (tt >= 0)
      acc = fmaf(cw[d * 4 + j], xz[(size_t)(b * SEQ + tt) * (2 * D_INNER) + d], acc);
  }
  xc[idx] = acc / (1.0f + __expf(-acc));
}

// ---------------- x_proj: ssm[row, 0..32] = xc[row,:] . x_proj_w[i,:] -------
__launch_bounds__(256)
__global__ void xproj_kernel(const float* __restrict__ xc,
                             const float* __restrict__ xw,   // (33, D_INNER)
                             float* __restrict__ ssm) {
  const int row  = blockIdx.x;                 // 0..NROWS-1
  const int lane = threadIdx.x & 31;
  const int wave = threadIdx.x >> 5;           // 0..7
  const float* xr = xc + (size_t)row * D_INNER;
  for (int i = wave; i < 2 * D_STATE + 1; i += 8) {
    const float* wr = xw + (size_t)i * D_INNER;
    float s = 0.0f;
    for (int d = lane; d < D_INNER; d += 32) s = fmaf(xr[d], wr[d], s);
    s += __shfl_xor(s, 16, 32);
    s += __shfl_xor(s, 8, 32);
    s += __shfl_xor(s, 4, 32);
    s += __shfl_xor(s, 2, 32);
    s += __shfl_xor(s, 1, 32);
    if (lane == 0) ssm[row * 33 + i] = s;
  }
}

// ---------------- selective scan: one lane per (d, n) state -----------------
__launch_bounds__(256)
__global__ void scan_kernel(const float* __restrict__ xc,
                            const float* __restrict__ ssm,
                            const float* __restrict__ log_A,
                            const float* __restrict__ dtw,   // (D_INNER)
                            const float* __restrict__ dtb,
                            const float* __restrict__ Dp,
                            float* __restrict__ y) {
  __shared__ float srow[33];
  const int tid = threadIdx.x;                 // 256: 16 channels x 16 states
  const int n   = tid & 15;
  const int dl  = tid >> 4;                    // 0..15
  const int b   = blockIdx.x >> 7;             // 128 blocks per batch
  const int d   = (blockIdx.x & 127) * 16 + dl;

  const float Aexp = __expf(log_A[d * D_STATE + n]);  // A = -exp(log_A)
  const float w_dt = dtw[d];
  const float b_dt = dtb[d];
  const float Dd   = Dp[d];
  float state = 0.0f;

  for (int t = 0; t < SEQ; ++t) {
    const int row = b * SEQ + t;
    __syncthreads();
    if (tid < 33) srow[tid] = ssm[row * 33 + tid];
    __syncthreads();
    float dr    = fmaf(srow[32], w_dt, b_dt);
    float delta = (dr > 20.0f) ? dr : __logf(1.0f + __expf(dr));
    delta = fminf(fmaxf(delta, 0.001f), 0.1f);
    float u  = xc[(size_t)row * D_INNER + d];
    float dA = __expf(-delta * Aexp);
    state = fmaf(dA, state, delta * srow[n] * u);
    float prod = srow[16 + n] * state;
    prod += __shfl_xor(prod, 1, 16);
    prod += __shfl_xor(prod, 2, 16);
    prod += __shfl_xor(prod, 4, 16);
    prod += __shfl_xor(prod, 8, 16);
    if (n == 0) y[(size_t)row * D_INNER + d] = fmaf(u, Dd, prod);
  }
}

// ---------------- LayerNorm + SiLU(z) gate + f16 pack -----------------------
__launch_bounds__(256)
__global__ void ln_gate_kernel(const float* __restrict__ y,
                               const float* __restrict__ xz,
                               const float* __restrict__ nw,
                               const float* __restrict__ nb,
                               _Float16* __restrict__ yzh) {
  __shared__ float r1[256];
  __shared__ float r2[256];
  const int row = blockIdx.x;
  const int tid = threadIdx.x;
  float s = 0.0f, sq = 0.0f;
  float vals[8];
  #pragma unroll
  for (int k = 0; k < 8; ++k) {
    float v = y[(size_t)row * D_INNER + tid + k * 256];
    vals[k] = v; s += v; sq += v * v;
  }
  r1[tid] = s; r2[tid] = sq;
  __syncthreads();
  for (int off = 128; off > 0; off >>= 1) {
    if (tid < off) { r1[tid] += r1[tid + off]; r2[tid] += r2[tid + off]; }
    __syncthreads();
  }
  const float mu  = r1[0] * (1.0f / D_INNER);
  const float var = r2[0] * (1.0f / D_INNER) - mu * mu;
  const float rs  = rsqrtf(var + 1e-5f);
  #pragma unroll
  for (int k = 0; k < 8; ++k) {
    const int d = tid + k * 256;
    float zv = xz[(size_t)row * (2 * D_INNER) + D_INNER + d];
    float g  = zv / (1.0f + __expf(-zv));
    float o  = fmaf((vals[k] - mu) * rs, nw[d], nb[d]) * g;
    yzh[(size_t)row * D_INNER + d] = (_Float16)o;
  }
}

// ---------------- launch ----------------------------------------------------
extern "C" void kernel_launch(void* const* d_in, const int* in_sizes, int n_in,
                              void* d_out, int out_size, void* d_ws, size_t ws_size,
                              hipStream_t stream) {
  (void)in_sizes; (void)n_in; (void)out_size; (void)ws_size;
  const float* x        = (const float*)d_in[0];
  const float* in_w     = (const float*)d_in[1];
  const float* conv_w   = (const float*)d_in[2];
  const float* conv_b   = (const float*)d_in[3];
  const float* x_proj_w = (const float*)d_in[4];
  const float* dt_w     = (const float*)d_in[5];
  const float* dt_b     = (const float*)d_in[6];
  const float* log_A    = (const float*)d_in[7];
  const float* D_param  = (const float*)d_in[8];
  const float* norm_w   = (const float*)d_in[9];
  const float* norm_b   = (const float*)d_in[10];
  const float* out_w    = (const float*)d_in[11];
  float* out = (float*)d_out;

  char* ws = (char*)d_ws;
  const size_t MB = 1024ull * 1024ull;
  _Float16* x_h    = (_Float16*)(ws + 0);          //  4 MB: 2048x1024 f16
  _Float16* win_h  = (_Float16*)(ws + 4 * MB);     //  8 MB: 4096x1024 f16
  _Float16* wout_h = (_Float16*)(ws + 12 * MB);    //  4 MB: 1024x2048 f16
  float*    xz     = (float*)   (ws + 16 * MB);    // 32 MB: 2048x4096 f32
  float*    xc     = (float*)   (ws + 48 * MB);    // 16 MB: 2048x2048 f32
  float*    ssm    = (float*)   (ws + 64 * MB);    // ~0.26 MB: 2048x33 f32
  float*    yv     = (float*)   (ws + 65 * MB);    // 16 MB: 2048x2048 f32
  _Float16* yz_h   = (_Float16*)(ws + 81 * MB);    //  8 MB: 2048x2048 f16

  // 1) f16 copies for WMMA GEMMs
  {
    int n = NROWS * D_MODEL;
    cvt_f16_kernel<<<(n + 255) / 256, 256, 0, stream>>>(x, x_h, n);
    n = (2 * D_INNER) * D_MODEL;
    cvt_f16_kernel<<<(n + 255) / 256, 256, 0, stream>>>(in_w, win_h, n);
    n = D_MODEL * D_INNER;
    cvt_f16_kernel<<<(n + 255) / 256, 256, 0, stream>>>(out_w, wout_h, n);
  }

  // 2) xz = x @ in_proj_w^T : M=2048, N=4096, K=1024
  {
    dim3 grid((2 * D_INNER) / BN, NROWS / BM);
    gemm_f16_wmma<2 * D_INNER, D_MODEL><<<grid, 256, 0, stream>>>(x_h, win_h, xz);
  }

  // 3) depthwise causal conv + SiLU -> xc
  {
    int n = BATCH * SEQ * D_INNER;
    conv_silu_kernel<<<(n + 255) / 256, 256, 0, stream>>>(xz, conv_w, conv_b, xc);
  }

  // 4) ssm = xc @ x_proj_w^T (33 outputs per row)
  xproj_kernel<<<NROWS, 256, 0, stream>>>(xc, x_proj_w, ssm);

  // 5) selective scan -> y (+ u*D)
  scan_kernel<<<BATCH * (D_INNER / 16), 256, 0, stream>>>(
      xc, ssm, log_A, dt_w, dt_b, D_param, yv);

  // 6) LayerNorm + SiLU(z) gate -> f16
  ln_gate_kernel<<<NROWS, 256, 0, stream>>>(yv, xz, norm_w, norm_b, yz_h);

  // 7) out = yz @ out_proj_w^T : M=2048, N=1024, K=2048
  {
    dim3 grid(D_MODEL / BN, NROWS / BM);
    gemm_f16_wmma<D_MODEL, D_INNER><<<grid, 256, 0, stream>>>(yz_h, wout_h, out);
  }
}
